// CotLayer_10969346474504
// MI455X (gfx1250) — matrix operand order repaired and checked by reading, not compiled, not run on hardware
//
#include <hip/hip_runtime.h>
#include <stdint.h>

// ---------------- problem constants ----------------
#define DIMC   256
#define HB     16          // batch
#define HH     56
#define WWI    56
#define PIXI   (HH*WWI)    // 3136
#define P_TOT  (HB*PIXI)   // 50176
#define EMBC   288
#define EMBPAD 320

typedef __attribute__((ext_vector_type(16))) __bf16 v16bf;
typedef __attribute__((ext_vector_type(8)))  __bf16 v8bf;
typedef __attribute__((ext_vector_type(8)))  float  v8f;

__device__ __forceinline__ unsigned short f2bf(float f) {
    union { float f; unsigned int u; } x; x.f = f;
    unsigned int u = x.u;
    return (unsigned short)((u + 0x7FFFu + ((u >> 16) & 1u)) >> 16);   // RNE
}
__device__ __forceinline__ unsigned int pk2(float lo, float hi) {     // two bf16 in a dword
    return (unsigned int)f2bf(lo) | ((unsigned int)f2bf(hi) << 16);
}

// ---------------- 0: x (NCHW f32) -> qkt bf16 [p][512], cols 0..255 ----------------
__global__ void k_transpose(const float* __restrict__ x, unsigned short* __restrict__ qkt) {
    __shared__ float tile[32][33];
    int pb = blockIdx.x * 32;
    int cb = blockIdx.y * 32;
    int tx = threadIdx.x & 31;          // pixel lane
    int ty = threadIdx.x >> 5;          // 8 channel slots
    int p  = pb + tx;
    int n  = p / PIXI, pix = p - n * PIXI;
    const float* xc = x + (size_t)n * DIMC * PIXI + pix;
#pragma unroll
    for (int j = 0; j < 4; ++j)
        tile[ty * 4 + j][tx] = xc[(size_t)(cb + ty * 4 + j) * PIXI];   // coalesced
    __syncthreads();
    int pl = threadIdx.x >> 3;
    int cj = (threadIdx.x & 7) * 4;
    uint2 v;
    v.x = pk2(tile[cj + 0][pl], tile[cj + 1][pl]);
    v.y = pk2(tile[cj + 2][pl], tile[cj + 3][pl]);
    *reinterpret_cast<uint2*>(qkt + (size_t)(pb + pl) * 512 + cb + cj) = v;
}

// zero the guard row (pixel index P_TOT) used for conv padding
__global__ void k_zeroguard(unsigned short* __restrict__ g) {
    g[blockIdx.x * 256 + threadIdx.x] = 0;
}

// ---------------- prep: weights -> bf16, BN folding ----------------
__global__ void k_wconv(const float* __restrict__ src, unsigned short* __restrict__ dst,
                        int rows, int rowsPad, int cin) {
    int i = blockIdx.x * 256 + threadIdx.x;
    if (i >= rowsPad * cin) return;
    int r = i / cin;
    dst[i] = (r < rows) ? f2bf(src[i]) : (unsigned short)0;
}

// ke_w [co][ci(64)][3][3] -> wke [co][tap*64+ci] bf16 (tap-major K)
__global__ void k_wke(const float* __restrict__ src, unsigned short* __restrict__ dst) {
    int i = blockIdx.x * 256 + threadIdx.x;
    if (i >= 256 * 576) return;
    int co = i / 576, kk = i % 576;
    int tap = kk >> 6, ci = kk & 63;
    dst[i] = f2bf(src[co * 576 + ci * 9 + tap]);
}

__global__ void k_bnprep(const float* __restrict__ g, const float* __restrict__ b,
                         const float* __restrict__ m, const float* __restrict__ v,
                         float* __restrict__ s, float* __restrict__ bias, int n) {
    int i = blockIdx.x * 256 + threadIdx.x;
    if (i >= n) return;
    float sc = g[i] * rsqrtf(v[i] + 1e-5f);
    s[i] = sc; bias[i] = b[i] - m[i] * sc;
}

// ---------------- key embed: grouped 3x3 conv via WMMA implicit GEMM ----------------
// block 256 thr = 8 waves; block: 64 out-ch (one group) x 64 pixels.
// Each wave: 16 co x 32 px (two accumulators share one A fragment).
// Pad handling: OOB lanes redirect their B-row pointer to the zero guard row
// (single v_cndmask_b32 on the pixel index, no per-element masking).
__global__ void k_keyembed(const unsigned short* __restrict__ qkt,
                           const unsigned short* __restrict__ wke,
                           const float* __restrict__ bns, const float* __restrict__ bnb,
                           float* __restrict__ kf, unsigned short* __restrict__ qktW) {
    int lane = threadIdx.x & 31;
    int wid  = threadIdx.x >> 5;
    int wco  = wid & 3, wpx = wid >> 2;
    int g      = blockIdx.y;
    int coBase = g * 64 + wco * 16;
    int pxBase = blockIdx.x * 64 + wpx * 32;
    int row    = lane & 15;
    int halfA  = (lane >> 4) * 16;
    int halfB  = (lane >> 4) * 16;
    const unsigned short* aRow = wke + (size_t)(coBase + row) * 576;
    // two pixel columns per lane (N tiles 0 and 1)
    int p0 = pxBase + row, p1 = p0 + 16;
    int n0 = p0 / PIXI, pix0 = p0 - n0 * PIXI;
    int n1 = p1 / PIXI, pix1 = p1 - n1 * PIXI;
    int h0 = pix0 / WWI, w0 = pix0 - h0 * WWI;
    int h1 = pix1 / WWI, w1 = pix1 - h1 * WWI;
    v8f acc0 = {}, acc1 = {};
#pragma unroll
    for (int tap = 0; tap < 9; ++tap) {
        int dy = tap / 3 - 1, dx = tap % 3 - 1;
        int hh0 = h0 + dy, ww0 = w0 + dx, hh1 = h1 + dy, ww1 = w1 + dx;
        bool ok0 = (hh0 >= 0) & (hh0 < HH) & (ww0 >= 0) & (ww0 < WWI);
        bool ok1 = (hh1 >= 0) & (hh1 < HH) & (ww1 >= 0) & (ww1 < WWI);
        int pi0 = ok0 ? (n0 * PIXI + hh0 * WWI + ww0) : P_TOT;   // guard row
        int pi1 = ok1 ? (n1 * PIXI + hh1 * WWI + ww1) : P_TOT;
        const unsigned short* bP0 = qkt + (size_t)pi0 * 512 + g * 64 + halfB;
        const unsigned short* bP1 = qkt + (size_t)pi1 * 512 + g * 64 + halfB;
#pragma unroll
        for (int kc = 0; kc < 64; kc += 32) {
            int kb = tap * 64 + kc;
            union { v16bf v; v8bf h2[2]; } A;
            A.h2[0] = *reinterpret_cast<const v8bf*>(aRow + kb + halfA);
            A.h2[1] = *reinterpret_cast<const v8bf*>(aRow + kb + halfA + 8);
            v16bf B0 = *reinterpret_cast<const v16bf*>(bP0 + kc);
            v16bf B1 = *reinterpret_cast<const v16bf*>(bP1 + kc);
            acc0 = __builtin_amdgcn_wmma_f32_16x16x32_bf16(
                false, A.v, false, B0, (short)0, acc0, false, false);
            acc1 = __builtin_amdgcn_wmma_f32_16x16x32_bf16(
                false, A.v, false, B1, (short)0, acc1, false, false);
        }
    }
    int pOutA = pxBase + (lane & 15);
    int pOutB = pOutA + 16;
    int coOff = (lane >> 4) * 8;
    float r0[8], r1[8];
#pragma unroll
    for (int v = 0; v < 8; ++v) {
        int co = coBase + coOff + v;
        float sA = bns[co], bA = bnb[co];
        r0[v] = fmaxf(sA * acc0[v] + bA, 0.f);
        r1[v] = fmaxf(sA * acc1[v] + bA, 0.f);
        kf[(size_t)co * P_TOT + pOutA] = r0[v];
        kf[(size_t)co * P_TOT + pOutB] = r1[v];
    }
    uint4 w0v = { pk2(r0[0], r0[1]), pk2(r0[2], r0[3]), pk2(r0[4], r0[5]), pk2(r0[6], r0[7]) };
    uint4 w1v = { pk2(r1[0], r1[1]), pk2(r1[2], r1[3]), pk2(r1[4], r1[5]), pk2(r1[6], r1[7]) };
    *reinterpret_cast<uint4*>(qktW + (size_t)pOutA * 512 + 256 + coBase + coOff) = w0v;
    *reinterpret_cast<uint4*>(qktW + (size_t)pOutB * 512 + 256 + coBase + coOff) = w1v;
}

// ---------------- generic 1x1 WMMA GEMM: out[co,p] = W[co,:] . act[p,:] ----------------
// Templated CIN -> fully unrolled K loop. Each wave: 16 co x 32 px (shared A frag).
// MODE: 0 = e1 (ReLU(BN) -> bf16 [p][dstBStride]); 1 = e2 (+bias -> f32 [co][P]);
//       2 = c1 (BN -> f32 [co][P])
template <int CIN, int MODE>
__global__ void k_gemm(const unsigned short* __restrict__ act, int actStride, int kOff,
                       const unsigned short* __restrict__ W,
                       const float* __restrict__ bns, const float* __restrict__ bnb,
                       int coValid,
                       float* __restrict__ dstF, unsigned short* __restrict__ dstB,
                       int dstBStride) {
    int lane = threadIdx.x & 31;
    int wid  = threadIdx.x >> 5;
    int wco  = wid & 3, wpx = wid >> 2;
    int coBase = blockIdx.y * 64 + wco * 16;
    int pxBase = blockIdx.x * 64 + wpx * 32;
    int row    = lane & 15;
    int halfA  = (lane >> 4) * 16;
    int halfB  = (lane >> 4) * 16;
    const unsigned short* aRow  = W + (size_t)(coBase + row) * CIN;
    const unsigned short* bRow0 = act + (size_t)(pxBase + row) * actStride + kOff;
    const unsigned short* bRow1 = act + (size_t)(pxBase + 16 + row) * actStride + kOff;
    __builtin_prefetch(bRow0, 0, 0);                 // global_prefetch_b8
    __builtin_prefetch(bRow1, 0, 0);
    v8f acc0 = {}, acc1 = {};
#pragma unroll
    for (int kb = 0; kb < CIN; kb += 32) {
        union { v16bf v; v8bf h2[2]; } A;
        A.h2[0] = *reinterpret_cast<const v8bf*>(aRow + kb + halfA);
        A.h2[1] = *reinterpret_cast<const v8bf*>(aRow + kb + halfA + 8);
        v16bf B0 = *reinterpret_cast<const v16bf*>(bRow0 + kb + halfB);
        v16bf B1 = *reinterpret_cast<const v16bf*>(bRow1 + kb + halfB);
        acc0 = __builtin_amdgcn_wmma_f32_16x16x32_bf16(
            false, A.v, false, B0, (short)0, acc0, false, false);
        acc1 = __builtin_amdgcn_wmma_f32_16x16x32_bf16(
            false, A.v, false, B1, (short)0, acc1, false, false);
    }
    int pA    = pxBase + (lane & 15);
    int pB    = pA + 16;
    int coOff = (lane >> 4) * 8;
    if (MODE == 0) {
        float r0[8], r1[8];
#pragma unroll
        for (int v = 0; v < 8; ++v) {
            int co = coBase + coOff + v;
            float s = bns[co], b = bnb[co];
            r0[v] = fmaxf(s * acc0[v] + b, 0.f);
            r1[v] = fmaxf(s * acc1[v] + b, 0.f);
        }
        uint4 w0v = { pk2(r0[0], r0[1]), pk2(r0[2], r0[3]), pk2(r0[4], r0[5]), pk2(r0[6], r0[7]) };
        uint4 w1v = { pk2(r1[0], r1[1]), pk2(r1[2], r1[3]), pk2(r1[4], r1[5]), pk2(r1[6], r1[7]) };
        *reinterpret_cast<uint4*>(dstB + (size_t)pA * dstBStride + coBase + coOff) = w0v;
        *reinterpret_cast<uint4*>(dstB + (size_t)pB * dstBStride + coBase + coOff) = w1v;
    } else {
#pragma unroll
        for (int v = 0; v < 8; ++v) {
            int co = coBase + coOff + v;
            float a0 = acc0[v], a1 = acc1[v];
            if (MODE == 1) {
                if (co < coValid) {
                    float b = bnb[co];                   // e2 bias
                    dstF[(size_t)co * P_TOT + pA] = a0 + b;
                    dstF[(size_t)co * P_TOT + pB] = a1 + b;
                }
            } else {
                float s = bns[co], b = bnb[co];
                dstF[(size_t)co * P_TOT + pA] = s * a0 + b;
                dstF[(size_t)co * P_TOT + pB] = s * a1 + b;
            }
        }
    }
}

// ---------------- GroupNorm stats: per (n, group of 9 ch) over 9*3136 values ----------------
__global__ void k_gnstats(const float* __restrict__ wemb2,
                          float* __restrict__ mu, float* __restrict__ rs) {
    int n = blockIdx.x >> 5, g = blockIdx.x & 31;
    float s = 0.f, s2 = 0.f;
    for (int i = threadIdx.x; i < 9 * PIXI; i += 256) {
        int ch = g * 9 + i / PIXI, pix = i % PIXI;
        float v = wemb2[(size_t)ch * P_TOT + n * PIXI + pix];
        s += v; s2 += v * v;
    }
    __shared__ float sh[256], sh2[256];
    sh[threadIdx.x] = s; sh2[threadIdx.x] = s2; __syncthreads();
    for (int o = 128; o > 0; o >>= 1) {
        if (threadIdx.x < o) { sh[threadIdx.x] += sh[threadIdx.x + o];
                               sh2[threadIdx.x] += sh2[threadIdx.x + o]; }
        __syncthreads();
    }
    if (threadIdx.x == 0) {
        float inv = 1.f / (9.f * PIXI);
        float m = sh[0] * inv, var = sh2[0] * inv - m * m;
        mu[blockIdx.x] = m; rs[blockIdx.x] = rsqrtf(var + 1e-5f);
    }
}

// ---------------- dynamic 9-tap conv + BN + swish -> y [c][p] ----------------
__global__ void k_dynconv(const float* __restrict__ wemb2, const float* __restrict__ xq,
                          const float* __restrict__ mu, const float* __restrict__ rs,
                          const float* __restrict__ gng, const float* __restrict__ gnb,
                          const float* __restrict__ b2s, const float* __restrict__ b2b,
                          float* __restrict__ y) {
    int idx = blockIdx.x * 256 + threadIdx.x;
    if (idx >= DIMC * P_TOT) return;
    int c = idx / P_TOT, p = idx - c * P_TOT;
    int cg = c >> 3;
    int n = p / PIXI, pix = p - n * PIXI;
    int h = pix / WWI, w = pix - h * WWI;
    int sg = n * 32 + cg;
    float m = mu[sg], r = rs[sg];
    float acc = 0.f;
#pragma unroll
    for (int tap = 0; tap < 9; ++tap) {
        int dy = tap / 3 - 1, dx = tap % 3 - 1;
        int hh = h + dy, ww = w + dx;
        if (hh < 0 || hh >= HH || ww < 0 || ww >= WWI) continue;
        int ch = cg * 9 + tap;
        float we = (wemb2[(size_t)ch * P_TOT + p] - m) * r * gng[ch] + gnb[ch];
        acc += we * xq[(size_t)c * P_TOT + n * PIXI + hh * WWI + ww];
    }
    float yv = b2s[c] * acc + b2b[c];
    y[idx] = yv / (1.f + __expf(-yv));                 // swish
}

// ---------------- global average pool of (y + k) ----------------
__global__ void k_gap(const float* __restrict__ y, const float* __restrict__ kf,
                      float* __restrict__ gap) {
    int n = blockIdx.x >> 8, c = blockIdx.x & 255;
    const float* yp = y  + (size_t)c * P_TOT + n * PIXI;
    const float* kp = kf + (size_t)c * P_TOT + n * PIXI;
    float s = 0.f;
    for (int i = threadIdx.x; i < PIXI; i += 256) s += yp[i] + kp[i];
    __shared__ float sh[256];
    sh[threadIdx.x] = s; __syncthreads();
    for (int o = 128; o > 0; o >>= 1) {
        if (threadIdx.x < o) sh[threadIdx.x] += sh[threadIdx.x + o];
        __syncthreads();
    }
    if (threadIdx.x == 0) gap[n * 256 + c] = sh[0] / (float)PIXI;
}

// ---------------- split attention MLP + radix-2 softmax ----------------
__global__ void k_attn(const float* __restrict__ gap,
                       const float* __restrict__ s1w, const float* __restrict__ s1b,
                       const float* __restrict__ s1s, const float* __restrict__ s1bias,
                       const float* __restrict__ s2w, const float* __restrict__ s2b,
                       float* __restrict__ att) {
    int n = blockIdx.x;
    __shared__ float gsh[256], a1[128];
    gsh[threadIdx.x] = gap[n * 256 + threadIdx.x];
    __syncthreads();
    if (threadIdx.x < 128) {
        float s = s1b[threadIdx.x];
        for (int ci = 0; ci < 256; ++ci) s += s1w[threadIdx.x * 256 + ci] * gsh[ci];
        a1[threadIdx.x] = fmaxf(s1s[threadIdx.x] * s + s1bias[threadIdx.x], 0.f);
    }
    __syncthreads();
    int c = threadIdx.x;
    float v0 = s2b[c * 2], v1 = s2b[c * 2 + 1];
    for (int j = 0; j < 128; ++j) {
        v0 += s2w[(c * 2) * 128 + j] * a1[j];
        v1 += s2w[(c * 2 + 1) * 128 + j] * a1[j];
    }
    float mx = fmaxf(v0, v1);
    float e0 = __expf(v0 - mx), e1 = __expf(v1 - mx);
    float inv = 1.f / (e0 + e1);
    att[(n * 256 + c) * 2]     = e0 * inv;
    att[(n * 256 + c) * 2 + 1] = e1 * inv;
}

// ---------------- final blend -> NCHW f32 ----------------
__global__ void k_out(const float* __restrict__ y, const float* __restrict__ kf,
                      const float* __restrict__ att, float* __restrict__ out) {
    int idx = blockIdx.x * 256 + threadIdx.x;
    if (idx >= DIMC * P_TOT) return;
    int n = idx / (DIMC * PIXI);
    int rem = idx - n * (DIMC * PIXI);
    int c = rem / PIXI, pix = rem - c * PIXI;
    int p = n * PIXI + pix;
    float a0 = att[(n * 256 + c) * 2], a1 = att[(n * 256 + c) * 2 + 1];
    out[idx] = y[(size_t)c * P_TOT + p] * a0 + kf[(size_t)c * P_TOT + p] * a1;
}

// ---------------- launch ----------------
extern "C" void kernel_launch(void* const* d_in, const int* in_sizes, int n_in,
                              void* d_out, int out_size, void* d_ws, size_t ws_size,
                              hipStream_t stream) {
    (void)in_sizes; (void)n_in; (void)out_size; (void)ws_size;
    const float* x    = (const float*)d_in[0];
    const float* ke_w = (const float*)d_in[1];
    const float* ke_g = (const float*)d_in[2];  const float* ke_b = (const float*)d_in[3];
    const float* ke_m = (const float*)d_in[4];  const float* ke_v = (const float*)d_in[5];
    const float* e1_w = (const float*)d_in[6];
    const float* e1_g = (const float*)d_in[7];  const float* e1_b = (const float*)d_in[8];
    const float* e1_m = (const float*)d_in[9];  const float* e1_v = (const float*)d_in[10];
    const float* e2_w = (const float*)d_in[11]; const float* e2_b = (const float*)d_in[12];
    const float* gng  = (const float*)d_in[13]; const float* gnb  = (const float*)d_in[14];
    const float* c1_w = (const float*)d_in[15];
    const float* c1_g = (const float*)d_in[16]; const float* c1_b = (const float*)d_in[17];
    const float* c1_m = (const float*)d_in[18]; const float* c1_v = (const float*)d_in[19];
    const float* b2_g = (const float*)d_in[20]; const float* b2_b = (const float*)d_in[21];
    const float* b2_m = (const float*)d_in[22]; const float* b2_v = (const float*)d_in[23];
    const float* s1_w = (const float*)d_in[24]; const float* s1_b = (const float*)d_in[25];
    const float* s1_g = (const float*)d_in[26]; const float* s1_bb= (const float*)d_in[27];
    const float* s1_m = (const float*)d_in[28]; const float* s1_v = (const float*)d_in[29];
    const float* s2_w = (const float*)d_in[30]; const float* s2_b = (const float*)d_in[31];
    float* out = (float*)d_out;

    // workspace carve (256B aligned)
    uint8_t* ws = (uint8_t*)d_ws;
    size_t off = 0;
    auto carve = [&](size_t bytes) {
        uint8_t* p = ws + off;
        off = (off + bytes + 255) & ~(size_t)255;
        return p;
    };
    // qkt has one extra zero "guard" row at pixel index P_TOT for conv padding
    unsigned short* qkt   = (unsigned short*)carve((size_t)(P_TOT + 1) * 512 * 2);
    unsigned short* wemb1 = (unsigned short*)carve((size_t)P_TOT * 128 * 2); // [p][128] bf16
    float* wemb2 = (float*)carve((size_t)EMBC * P_TOT * 4);                  // [288][P]
    float* xq    = (float*)carve((size_t)DIMC * P_TOT * 4);                  // [256][P]
    float* yb    = (float*)carve((size_t)DIMC * P_TOT * 4);                  // [256][P]
    float* kf    = (float*)carve((size_t)DIMC * P_TOT * 4);                  // [256][P]
    unsigned short* wke = (unsigned short*)carve((size_t)256 * 576 * 2);
    unsigned short* we1 = (unsigned short*)carve((size_t)128 * 512 * 2);
    unsigned short* we2 = (unsigned short*)carve((size_t)EMBPAD * 128 * 2);
    unsigned short* wc1 = (unsigned short*)carve((size_t)256 * 256 * 2);
    float* bn = (float*)carve(2048 * 4);
    float* ke_s = bn;        float* ke_bb = bn + 256;
    float* e1_s = bn + 512;  float* e1_bb = bn + 640;
    float* c1_s = bn + 768;  float* c1_bb = bn + 1024;
    float* b2_s = bn + 1280; float* b2_bb = bn + 1536;
    float* s1_s = bn + 1792; float* s1_bs = bn + 1920;
    float* mu  = (float*)carve(512 * 4);
    float* rsb = (float*)carve(512 * 4);
    float* gap = (float*)carve(HB * 256 * 4);
    float* att = (float*)carve(HB * 512 * 4);

    // 1) transpose x to bf16 [p][512] (cols 0..255); zero the pad guard row
    k_transpose<<<dim3(P_TOT / 32, DIMC / 32), 256, 0, stream>>>(x, qkt);
    k_zeroguard<<<2, 256, 0, stream>>>(qkt + (size_t)P_TOT * 512);
    // 2) weight repacks + BN folding
    k_wke<<<(256 * 576 + 255) / 256, 256, 0, stream>>>(ke_w, wke);
    k_wconv<<<(128 * 512 + 255) / 256, 256, 0, stream>>>(e1_w, we1, 128, 128, 512);
    k_wconv<<<(EMBPAD * 128 + 255) / 256, 256, 0, stream>>>(e2_w, we2, EMBC, EMBPAD, 128);
    k_wconv<<<(256 * 256 + 255) / 256, 256, 0, stream>>>(c1_w, wc1, 256, 256, 256);
    k_bnprep<<<1, 256, 0, stream>>>(ke_g, ke_b, ke_m, ke_v, ke_s, ke_bb, 256);
    k_bnprep<<<1, 256, 0, stream>>>(e1_g, e1_b, e1_m, e1_v, e1_s, e1_bb, 128);
    k_bnprep<<<1, 256, 0, stream>>>(c1_g, c1_b, c1_m, c1_v, c1_s, c1_bb, 256);
    k_bnprep<<<1, 256, 0, stream>>>(b2_g, b2_b, b2_m, b2_v, b2_s, b2_bb, 256);
    k_bnprep<<<1, 256, 0, stream>>>(s1_g, s1_bb, s1_m, s1_v, s1_s, s1_bs, 128);
    // 3) key embed: grouped 3x3 conv (WMMA) + BN + ReLU -> kf, qkt cols 256..511
    k_keyembed<<<dim3(P_TOT / 64, 4), 256, 0, stream>>>(qkt, wke, ke_s, ke_bb, kf, qkt);
    // 4) e1: 512->128, ReLU(BN) -> wemb1 bf16
    k_gemm<512, 0><<<dim3(P_TOT / 64, 2), 256, 0, stream>>>(
        qkt, 512, 0, we1, e1_s, e1_bb, 128, nullptr, wemb1, 128);
    // 5) e2: 128->288, +bias -> wemb2 f32
    k_gemm<128, 1><<<dim3(P_TOT / 64, 5), 256, 0, stream>>>(
        wemb1, 128, 0, we2, nullptr, e2_b, EMBC, wemb2, nullptr, 0);
    // 6) c1: 256->256 on x half of qkt, BN -> xq f32
    k_gemm<256, 2><<<dim3(P_TOT / 64, 4), 256, 0, stream>>>(
        qkt, 512, 0, wc1, c1_s, c1_bb, 256, xq, nullptr, 0);
    // 7) GroupNorm stats (32 groups x 16 images)
    k_gnstats<<<512, 256, 0, stream>>>(wemb2, mu, rsb);
    // 8) dynamic conv + BN + swish -> y
    k_dynconv<<<(DIMC * P_TOT) / 256, 256, 0, stream>>>(wemb2, xq, mu, rsb, gng, gnb,
                                                        b2_s, b2_bb, yb);
    // 9) GAP of (y + k)
    k_gap<<<HB * 256, 256, 0, stream>>>(yb, kf, gap);
    // 10) split-attention MLP + softmax
    k_attn<<<HB, 256, 0, stream>>>(gap, s1_w, s1_b, s1_s, s1_bs, s2_w, s2_b, att);
    // 11) blend -> NCHW output
    k_out<<<(DIMC * P_TOT) / 256, 256, 0, stream>>>(yb, kf, att, out);
}
